// SelectiveSSM_82179904242171
// MI455X (gfx1250) — matrix operand order repaired and checked by reading, not compiled, not run on hardware
//
#include <hip/hip_runtime.h>
#include <stdint.h>

// ---------------------------------------------------------------------------
// Selective SSM (Mamba-style) forward for MI455X (gfx1250, wave32, WMMA).
//   B=4, S=2048, D_MODEL=2048, D_INNER=4096, D_STATE=16, D_CONV=4
// GEMMs run in bf16 via v_wmma_f32_16x16x32_bf16 with fp32 accumulation,
// LDS double-buffered with global_load_async_to_lds_b128 (ASYNCcnt pipeline).
// ---------------------------------------------------------------------------

#define BATCH   4
#define SEQ     2048
#define DMODEL  2048
#define DINNER  4096
#define DSTATE  16
#define NROWS   (BATCH * SEQ)          // 8192 "M" rows
#define N1      (2 * DINNER)           // 8192
#define K1      DMODEL                 // 2048
#define N2      DMODEL                 // 2048
#define K2      DINNER                 // 4096

typedef __attribute__((ext_vector_type(16))) __bf16 v16bf;
typedef __attribute__((ext_vector_type(8)))  float  v8f;
typedef int vi4 __attribute__((vector_size(4 * sizeof(int))));

union FragBF { v16bf v; uint4 q[2]; };

// ---- workspace layout (bytes) ---------------------------------------------
#define WS_WIN_BF   ((size_t)0)                         // 8192*2048 bf16 = 32MB
#define WS_X_BF     (WS_WIN_BF  + (size_t)33554432)     // 8192*2048 bf16 = 32MB
#define WS_XZ       (WS_X_BF    + (size_t)33554432)     // 8192*8192 bf16 = 128MB
#define WS_XCONV    (WS_XZ      + (size_t)134217728)    // 8192*4096 bf16 = 64MB
#define WS_OUTW_BF  (WS_XCONV   + (size_t)67108864)     // 2048*4096 bf16 = 16MB
#define WS_ABAR     (WS_OUTW_BF + (size_t)16777216)     // 8192*16 f32
#define WS_DTB      (WS_ABAR    + (size_t)524288)
#define WS_CMAT     (WS_DTB     + (size_t)524288)
#define WS_YG       (WS_CMAT    + (size_t)524288)       // 8192*4096 bf16 = 64MB
#define WS_MEAN     (WS_YG      + (size_t)67108864)     // 1 f32

// ---- async global->LDS (guarded; fall back to sync copy) ------------------
#if defined(__gfx1250__) && __has_builtin(__builtin_amdgcn_global_load_async_to_lds_b128)
#define HAVE_ASYNC 1
#else
#define HAVE_ASYNC 0
#endif

__device__ __forceinline__ void copy16_to_lds(const __bf16* g, unsigned char* l) {
#if HAVE_ASYNC
  __builtin_amdgcn_global_load_async_to_lds_b128(
      (__attribute__((address_space(1))) vi4*)(g),
      (__attribute__((address_space(3))) vi4*)(l), 0, 0);
#else
  *(uint4*)l = *(const uint4*)g;
#endif
}

__device__ __forceinline__ void wait_lds_copies() {
#if HAVE_ASYNC
#if __has_builtin(__builtin_amdgcn_s_wait_asynccnt)
  __builtin_amdgcn_s_wait_asynccnt(0);
#else
  asm volatile("s_wait_asynccnt 0" ::: "memory");
#endif
#endif
}

__device__ __forceinline__ float silu_f(float v) { return v / (1.0f + __expf(-v)); }

__device__ __forceinline__ void store_c(float* p, float v)  { *p = v; }
__device__ __forceinline__ void store_c(__bf16* p, float v) { *p = (__bf16)v; }

// ---------------------------------------------------------------------------
// fp32 -> bf16 conversion (4 elems/thread; all sizes divisible by 4)
// ---------------------------------------------------------------------------
__global__ __launch_bounds__(256) void f2bf_kernel(const float* __restrict__ in,
                                                   __bf16* __restrict__ out, int n) {
  int i = (blockIdx.x * 256 + threadIdx.x) * 4;
  if (i + 3 < n) {
    float4 v = *(const float4*)(in + i);
    out[i + 0] = (__bf16)v.x; out[i + 1] = (__bf16)v.y;
    out[i + 2] = (__bf16)v.z; out[i + 3] = (__bf16)v.w;
  }
}

// ---------------------------------------------------------------------------
// mean(dt_bias) over 4096 elems -> scalar
// ---------------------------------------------------------------------------
__global__ __launch_bounds__(256) void meanbias_kernel(const float* __restrict__ dt_bias,
                                                       float* __restrict__ out) {
  __shared__ float red[256];
  float s = 0.0f;
  for (int i = threadIdx.x; i < DINNER; i += 256) s += dt_bias[i];
  red[threadIdx.x] = s;
  __syncthreads();
  for (int off = 128; off > 0; off >>= 1) {
    if (threadIdx.x < off) red[threadIdx.x] += red[threadIdx.x + off];
    __syncthreads();
  }
  if (threadIdx.x == 0) out[0] = red[0] * (1.0f / (float)DINNER);
}

// ---------------------------------------------------------------------------
// bf16 GEMM: C[M,N] = A[M,K] * Bw[N,K]^T   (both row-major along K)
// Block tile 256x128, 256 threads = 8 waves (4x2), wave tile 64x64
// (16 v_wmma per K-slab per wave). K stepped by 32.  LDS is double-buffered:
// async copies for slab kt+1 are issued right after the barrier and overlap
// the 16 WMMAs of slab kt; one barrier + one s_wait_asynccnt per slab.
// ---------------------------------------------------------------------------
#define ROWB 80                 // LDS row stride: 32 bf16 (64B) + 16B pad
#define ASZ  (256 * ROWB)       // 20480 B per A buffer
#define BSZ  (128 * ROWB)       // 10240 B per B buffer

template <typename COUT>
__global__ __launch_bounds__(256) void gemm_bf16_kernel(
    const __bf16* __restrict__ A, const __bf16* __restrict__ Bw,
    COUT* __restrict__ C, int M, int N, int K) {
  __shared__ __align__(16) unsigned char smA[2 * ASZ];   // 40 KB
  __shared__ __align__(16) unsigned char smB[2 * BSZ];   // 20 KB

  const int tid  = threadIdx.x;
  const int m0   = blockIdx.x * 256;
  const int n0   = blockIdx.y * 128;
  const int wid  = tid >> 5;
  const int lane = tid & 31;
  const int wM   = wid >> 1;      // 0..3 -> 64-row band
  const int wN   = wid & 1;       // 0..1 -> 64-col band
  const int lh   = lane >> 4;     // lane half
  const int l16  = lane & 15;

  v8f acc[4][4];
#pragma unroll
  for (int mi = 0; mi < 4; ++mi)
#pragma unroll
    for (int ni = 0; ni < 4; ++ni) acc[mi][ni] = (v8f)0.0f;

  // per-thread staging map: A = 256 rows x 64B (1 row/thread, 4x b128);
  //                         B = 128 rows x 64B (half row/thread, 2x b128)
  const __bf16* gAc = A  + (size_t)(m0 + tid) * K;
  const __bf16* gBc = Bw + (size_t)(n0 + (tid >> 1)) * K + (tid & 1) * 16;
  unsigned char* lAr = smA + tid * ROWB;
  unsigned char* lBr = smB + (tid >> 1) * ROWB + (tid & 1) * 32;

  // prologue: stage slab 0 into buffer 0
  copy16_to_lds(gAc,      lAr);
  copy16_to_lds(gAc + 8,  lAr + 16);
  copy16_to_lds(gAc + 16, lAr + 32);
  copy16_to_lds(gAc + 24, lAr + 48);
  copy16_to_lds(gBc,      lBr);
  copy16_to_lds(gBc + 8,  lBr + 16);
  gAc += 32; gBc += 32;

  const int nk = K >> 5;
  for (int kt = 0; kt < nk; ++kt) {
    const int cur = kt & 1;
    wait_lds_copies();      // slab kt landed (issued one stage earlier)
    __syncthreads();        // visible to all waves; buffer cur^1 reads done

    if (kt + 1 < nk) {      // overlap: stage slab kt+1 into the other buffer
      __builtin_prefetch(gAc + 32, 0, 1);
      __builtin_prefetch(gBc + 32, 0, 1);
      const int nxt = cur ^ 1;
      unsigned char* a = lAr + nxt * ASZ;
      unsigned char* b = lBr + nxt * BSZ;
      copy16_to_lds(gAc,      a);
      copy16_to_lds(gAc + 8,  a + 16);
      copy16_to_lds(gAc + 16, a + 32);
      copy16_to_lds(gAc + 24, a + 48);
      copy16_to_lds(gBc,      b);
      copy16_to_lds(gBc + 8,  b + 16);
      gAc += 32; gBc += 32;
    }

    const unsigned char* bA = smA + cur * ASZ;
    const unsigned char* bB = smB + cur * BSZ;

    FragBF aF[4], bF[4];
#pragma unroll
    for (int mi = 0; mi < 4; ++mi) {
      const int r = wM * 64 + mi * 16 + l16;
      // A 16x32 bf16: lanes<16 hold K 0..7 / 16..23, lanes>=16 hold K 8..15 / 24..31
      aF[mi].q[0] = *(const uint4*)(bA + r * ROWB + lh * 16);
      aF[mi].q[1] = *(const uint4*)(bA + r * ROWB + 32 + lh * 16);
    }
#pragma unroll
    for (int ni = 0; ni < 4; ++ni) {
      const int r = wN * 64 + ni * 16 + l16;
      // B 32x16 bf16: lane = N, lanes<16 hold K 0..7 / 16..23, lanes>=16 hold K 8..15 / 24..31
      bF[ni].q[0] = *(const uint4*)(bB + r * ROWB + lh * 16);
      bF[ni].q[1] = *(const uint4*)(bB + r * ROWB + 32 + lh * 16);
    }

#pragma unroll
    for (int mi = 0; mi < 4; ++mi)
#pragma unroll
      for (int ni = 0; ni < 4; ++ni)
        acc[mi][ni] = __builtin_amdgcn_wmma_f32_16x16x32_bf16(
            false, aF[mi].v, false, bF[ni].v, (short)0, acc[mi][ni], false, false);
  }

  // epilogue: C/D layout — lane(0..15)=N col, vgpr e -> row (lh*8 + e)
#pragma unroll
  for (int mi = 0; mi < 4; ++mi) {
#pragma unroll
    for (int ni = 0; ni < 4; ++ni) {
      const int row0 = m0 + wM * 64 + mi * 16 + lh * 8;
      const int col  = n0 + wN * 64 + ni * 16 + l16;
#pragma unroll
      for (int e = 0; e < 8; ++e)
        store_c(&C[(size_t)(row0 + e) * N + col], acc[mi][ni][e]);
    }
  }
}

// ---------------------------------------------------------------------------
// depthwise causal conv (width 4) + SiLU on x_ssm half of xz
// ---------------------------------------------------------------------------
__global__ __launch_bounds__(256) void conv_silu_kernel(
    const __bf16* __restrict__ xz, const float* __restrict__ conv_w,
    const float* __restrict__ conv_b, __bf16* __restrict__ xconv) {
  const int idx = blockIdx.x * 256 + threadIdx.x;   // (b, t, c) with c fastest
  const int c = idx & (DINNER - 1);
  const int t = (idx >> 12) & (SEQ - 1);
  const int b = idx >> 23;
  float s = conv_b[c];
#pragma unroll
  for (int j = 0; j < 4; ++j) {
    const int tt = t + j - 3;
    if (tt >= 0)
      s += (float)xz[((size_t)(b * SEQ + tt)) * N1 + c] * conv_w[c * 4 + j];
  }
  xconv[idx] = (__bf16)silu_f(s);
}

// ---------------------------------------------------------------------------
// p = x_conv @ W_x^T per (b,t); emit A_bar = exp(A*dt), dt*B, C (16 each).
// One block per (b,t); stage the 4096-wide row in LDS; each of 8 waves
// reduces ~4 of the 33 outputs with shuffle reduction (no extra barriers).
// ---------------------------------------------------------------------------
__global__ __launch_bounds__(256) void proj_params_kernel(
    const __bf16* __restrict__ xconv, const float* __restrict__ W_x,
    const float* __restrict__ A_log, const float* __restrict__ meanb,
    float* __restrict__ abar, float* __restrict__ dtb, float* __restrict__ cmat) {
  __shared__ float xs[DINNER];
  __shared__ float pj[33];
  const int bt = blockIdx.x;
  for (int c = threadIdx.x; c < DINNER; c += 256)
    xs[c] = (float)xconv[(size_t)bt * DINNER + c];
  __syncthreads();

  const int wid = threadIdx.x >> 5;
  const int lane = threadIdx.x & 31;
  for (int j = wid; j < 33; j += 8) {
    const float* w = W_x + (size_t)j * DINNER;
    float s = 0.0f;
    for (int c = lane; c < DINNER; c += 32) s += xs[c] * w[c];
#pragma unroll
    for (int off = 16; off > 0; off >>= 1) s += __shfl_down(s, off, 32);
    if (lane == 0) pj[j] = s;
  }
  __syncthreads();

  if (threadIdx.x < DSTATE) {
    const int st = threadIdx.x;
    const float raw = pj[32] + meanb[0];
    const float dt = (raw > 20.0f) ? raw : log1pf(__expf(raw));  // softplus
    const float A = -__expf(A_log[st]);   // A_log is channel-independent (row 0)
    abar[bt * DSTATE + st] = __expf(A * dt);
    dtb [bt * DSTATE + st] = dt * pj[st];
    cmat[bt * DSTATE + st] = pj[DSTATE + st];
  }
}

// ---------------------------------------------------------------------------
// Sequential scan over SEQ. One lane per channel, h[16] in registers.
// A_bar/dtB/C are uniform per (b,t) -> scalar-loadable broadcasts.
// Fuses the +D*x_conv and *silu(z) gating; output bf16 for GEMM2.
// ---------------------------------------------------------------------------
__global__ __launch_bounds__(256) void scan_kernel(
    const __bf16* __restrict__ xconv, const __bf16* __restrict__ xz,
    const float* __restrict__ abar, const float* __restrict__ dtb,
    const float* __restrict__ cmat, const float* __restrict__ Dw,
    __bf16* __restrict__ yg) {
  const int b = blockIdx.x >> 4;
  const int c = ((blockIdx.x & 15) << 8) + threadIdx.x;
  const float Dc = Dw[c];
  float h[DSTATE];
#pragma unroll
  for (int s = 0; s < DSTATE; ++s) h[s] = 0.0f;

  for (int t = 0; t < SEQ; ++t) {
    const size_t bt = (size_t)b * SEQ + t;
    const float xc = (float)xconv[bt * DINNER + c];
    const float* ab = abar + bt * DSTATE;
    const float* db = dtb + bt * DSTATE;
    const float* cm = cmat + bt * DSTATE;
    float y = 0.0f;
#pragma unroll
    for (int s = 0; s < DSTATE; ++s) {
      h[s] = ab[s] * h[s] + db[s] * xc;
      y += h[s] * cm[s];
    }
    const float z = (float)xz[bt * N1 + DINNER + c];
    yg[bt * DINNER + c] = (__bf16)((y + Dc * xc) * silu_f(z));
  }
}

// ---------------------------------------------------------------------------
extern "C" void kernel_launch(void* const* d_in, const int* in_sizes, int n_in,
                              void* d_out, int out_size, void* d_ws, size_t ws_size,
                              hipStream_t stream) {
  const float* x       = (const float*)d_in[0];
  const float* W_in    = (const float*)d_in[1];
  const float* conv_w  = (const float*)d_in[2];
  const float* conv_b  = (const float*)d_in[3];
  const float* W_x     = (const float*)d_in[4];
  const float* A_log   = (const float*)d_in[5];
  const float* dt_bias = (const float*)d_in[6];
  const float* out_W   = (const float*)d_in[7];
  const float* Dw      = (const float*)d_in[8];
  (void)in_sizes; (void)n_in; (void)out_size; (void)ws_size;

  char* ws = (char*)d_ws;
  __bf16* win_bf  = (__bf16*)(ws + WS_WIN_BF);
  __bf16* x_bf    = (__bf16*)(ws + WS_X_BF);
  __bf16* xz_bf   = (__bf16*)(ws + WS_XZ);
  __bf16* xconv   = (__bf16*)(ws + WS_XCONV);
  __bf16* outw_bf = (__bf16*)(ws + WS_OUTW_BF);
  float*  abar    = (float*)(ws + WS_ABAR);
  float*  dtb     = (float*)(ws + WS_DTB);
  float*  cmat    = (float*)(ws + WS_CMAT);
  __bf16* yg      = (__bf16*)(ws + WS_YG);
  float*  meanb   = (float*)(ws + WS_MEAN);

  // 1) fp32 -> bf16 (x, W_in, out_W)
  f2bf_kernel<<<(N1 * K1) / 1024, 256, 0, stream>>>(W_in, win_bf, N1 * K1);
  f2bf_kernel<<<(NROWS * K1) / 1024, 256, 0, stream>>>(x, x_bf, NROWS * K1);
  f2bf_kernel<<<(N2 * K2) / 1024, 256, 0, stream>>>(out_W, outw_bf, N2 * K2);

  // 2) mean(dt_bias)
  meanbias_kernel<<<1, 256, 0, stream>>>(dt_bias, meanb);

  // 3) xz = x @ W_in^T  (8192 x 8192, K=2048)
  gemm_bf16_kernel<__bf16><<<dim3(NROWS / 256, N1 / 128), 256, 0, stream>>>(
      x_bf, win_bf, xz_bf, NROWS, N1, K1);

  // 4) depthwise conv + SiLU
  conv_silu_kernel<<<(BATCH * SEQ * DINNER) / 256, 256, 0, stream>>>(
      xz_bf, conv_w, conv_b, xconv);

  // 5) p-projection -> A_bar, dt*B, C per (b,t)
  proj_params_kernel<<<BATCH * SEQ, 256, 0, stream>>>(
      xconv, W_x, A_log, meanb, abar, dtb, cmat);

  // 6) selective scan + gating
  scan_kernel<<<(BATCH * DINNER) / 256, 256, 0, stream>>>(
      xconv, xz_bf, abar, dtb, cmat, Dw, yg);

  // 7) out = y_gated @ out_W^T  (8192 x 2048, K=4096), fp32 output
  gemm_bf16_kernel<float><<<dim3(NROWS / 256, N2 / 128), 256, 0, stream>>>(
      yg, outw_bf, (float*)d_out, NROWS, N2, K2);
}
